// CustomMultiheadAttention_21234318312014
// MI455X (gfx1250) — matrix operand order repaired and checked by reading, not compiled, not run on hardware
//
#include <hip/hip_runtime.h>
#include <hip/hip_bf16.h>

namespace {

constexpr int Bc  = 4;
constexpr int Tc  = 2048;
constexpr int Sc  = 2048;
constexpr int Ec  = 1024;
constexpr int Hc  = 16;
constexpr int HDc = 64;
constexpr int Mc  = Bc * Tc;   // 8192 rows

typedef __attribute__((ext_vector_type(16))) __bf16 v16bf;
typedef __attribute__((ext_vector_type(8)))  __bf16 v8bf;
typedef __attribute__((ext_vector_type(8)))  float  v8f;

__device__ __forceinline__ v8f wmma_bf16(v16bf a, v16bf b, v8f c) {
  // (neg_a, A, neg_b, B, c_mod, C, reuse_a, reuse_b)
  return __builtin_amdgcn_wmma_f32_16x16x32_bf16(false, a, false, b, (short)0, c,
                                                 false, false);
}

__device__ __forceinline__ v16bf join16(v8bf lo, v8bf hi) {
  v16bf r;
#pragma unroll
  for (int i = 0; i < 8; ++i) { r[i] = lo[i]; r[i + 8] = hi[i]; }
  return r;
}

// DPP16 lane shuffle (wave32: DPP "row" = 16 lanes = one half-wave).
template <int CTRL>
__device__ __forceinline__ float dppf(float x) {
  return __int_as_float(__builtin_amdgcn_update_dpp(
      __float_as_int(x), __float_as_int(x), CTRL, 0xF, 0xF, false));
}

// All-reduce max across each 16-lane half: xor1, xor2 via quad_perm, then
// quad-level all-reduce via row_ror:4 and row_ror:8. Pure VALU, no LDS.
__device__ __forceinline__ float rowmax16(float x) {
  x = fmaxf(x, dppf<0xB1>(x));    // quad_perm [1,0,3,2]  (xor 1)
  x = fmaxf(x, dppf<0x4E>(x));    // quad_perm [2,3,0,1]  (xor 2)
  x = fmaxf(x, dppf<0x124>(x));   // row_ror:4
  x = fmaxf(x, dppf<0x128>(x));   // row_ror:8
  return x;
}

// Y = outScale * (X @ W^T + bias)
// OUT_MODE 0: bf16 row-major [M, E]
// OUT_MODE 1: bf16 head-transposed [B, H, hd, T]  (for V^T)
// OUT_MODE 2: f32  row-major [M, E]               (final output)
template <typename XT, int OUT_MODE>
__global__ __launch_bounds__(256) void proj_kernel(
    const XT* __restrict__ X, const float* __restrict__ W,
    const float* __restrict__ bias, void* __restrict__ Yv, float outScale) {
  const int lane   = threadIdx.x & 31;
  const int wave   = threadIdx.x >> 5;
  const int laneLo = lane & 15;
  const int laneHi = lane >> 4;
  const int tilesN = Ec / 16;               // 64
  const int tile   = blockIdx.x * 8 + wave; // one 16x16 tile per wave
  const int tm     = tile / tilesN;
  const int tn     = tile % tilesN;

  const size_t xRow = (size_t)(tm * 16 + laneLo) * Ec;   // A: lane = M
  const size_t wRow = (size_t)(tn * 16 + laneLo) * Ec;   // B: lane = N (row of W)

  auto loadA = [&](int kb) -> v16bf {
    if constexpr (sizeof(XT) == 4) {        // f32 input: load + convert to bf16
      const float* xp = (const float*)X + xRow + kb + laneHi * 8;
      float xr[16];
      *(float4*)(xr + 0)  = *(const float4*)(xp + 0);   // K = laneHi*8 + 0..7
      *(float4*)(xr + 4)  = *(const float4*)(xp + 4);
      *(float4*)(xr + 8)  = *(const float4*)(xp + 16);  // K = 16 + laneHi*8 + 0..7
      *(float4*)(xr + 12) = *(const float4*)(xp + 20);
      v16bf t;
#pragma unroll
      for (int i = 0; i < 16; ++i) t[i] = (__bf16)xr[i];
      return t;
    } else {                                // bf16 input (ctx @ Wo^T)
      const __bf16* xp = (const __bf16*)X + xRow + kb + laneHi * 8;
      return join16(*(const v8bf*)xp, *(const v8bf*)(xp + 16));
    }
  };
  auto loadB = [&](int kb) -> v16bf {
    // B-frag: lane n holds W[n][kb + laneHi*8 + 0..7] and W[n][kb + 16 + laneHi*8 + 0..7]
    const float* wp = W + wRow + kb + laneHi * 8;
    float wr[16];
    *(float4*)(wr + 0)  = *(const float4*)(wp + 0);
    *(float4*)(wr + 4)  = *(const float4*)(wp + 4);
    *(float4*)(wr + 8)  = *(const float4*)(wp + 16);
    *(float4*)(wr + 12) = *(const float4*)(wp + 20);
    v16bf t;
#pragma unroll
    for (int i = 0; i < 16; ++i) t[i] = (__bf16)wr[i];
    return t;
  };

  // software pipeline (2x unrolled ping-pong so the swap is free)
  v8f acc = {};
  v16bf a0 = loadA(0), b0 = loadB(0);
  v16bf a1, b1;
  for (int kb = 0; kb < Ec; kb += 64) {
    a1 = loadA(kb + 32); b1 = loadB(kb + 32);
    acc = wmma_bf16(a0, b0, acc);
    if (kb + 64 < Ec) { a0 = loadA(kb + 64); b0 = loadB(kb + 64); }
    acc = wmma_bf16(a1, b1, acc);
  }

  const float bb = bias[tn * 16 + laneLo];
#pragma unroll
  for (int r = 0; r < 8; ++r) {
    const int mrow = tm * 16 + laneHi * 8 + r;   // C/D: VGPR r -> row laneHi*8+r
    const int ncol = tn * 16 + laneLo;           //       lane  -> col
    const float y  = (acc[r] + bb) * outScale;
    if constexpr (OUT_MODE == 0) {
      ((__bf16*)Yv)[(size_t)mrow * Ec + ncol] = (__bf16)y;
    } else if constexpr (OUT_MODE == 1) {
      const int bi = mrow / Tc, t = mrow % Tc;
      const int h  = ncol >> 6, d = ncol & 63;
      ((__bf16*)Yv)[(((size_t)bi * Hc + h) * HDc + d) * Sc + t] = (__bf16)y;
    } else {
      ((float*)Yv)[(size_t)mrow * Ec + ncol] = y;
    }
  }
}

// Flash attention: one wave per (b, h, 16-row query block), streaming 32 keys/iter.
__global__ __launch_bounds__(128) void attn_kernel(
    const __bf16* __restrict__ Q, const __bf16* __restrict__ K,
    const __bf16* __restrict__ VT, __bf16* __restrict__ O) {
  __shared__ __bf16 Pbuf[4][16][32];     // per-wave P transpose staging
  const int lane   = threadIdx.x & 31;
  const int wave   = threadIdx.x >> 5;
  const int laneLo = lane & 15;
  const int laneHi = lane >> 4;
  const int wid = blockIdx.x * 4 + wave;       // 0 .. 8191
  const int qb  = wid & (Tc / 16 - 1);
  const int h   = (wid >> 7) & (Hc - 1);
  const int b   = wid >> 11;

  // Q A-frags for the two hd chunks (K = 0..31, 32..63). Scale already folded in.
  v16bf aq[2];
  {
    const __bf16* qp = Q + ((size_t)b * Tc + qb * 16 + laneLo) * Ec + h * HDc;
#pragma unroll
    for (int kk = 0; kk < 2; ++kk) {
      const __bf16* p = qp + kk * 32 + laneHi * 16;
      aq[kk] = join16(*(const v8bf*)p, *(const v8bf*)(p + 8));
    }
  }
  // all-ones B-frag: P @ ones gives row sums in accumulator layout
  v16bf ones;
#pragma unroll
  for (int i = 0; i < 16; ++i) ones[i] = (__bf16)1.0f;

  // K B-frag base: lane = key col, contiguous hd slice at + laneHi*16
  const size_t kRow = ((size_t)b * Sc + laneLo) * Ec + h * HDc + laneHi * 16;
  // V^T B-frag base: lane = hd col, contiguous key slice at + laneHi*16
  const size_t vRow = (((size_t)b * Hc + h) * HDc + laneLo) * Sc + laneHi * 16;

  float mrow[8], lrow[8];
  v8f accd[4] = {};
#pragma unroll
  for (int r = 0; r < 8; ++r) { mrow[r] = -3.0e38f; lrow[r] = 0.0f; }

  // load the four K B-frags for key block starting at sb
  auto loadK = [&](int sb, v16bf bk[4]) {
    const __bf16* kb0 = K + kRow + (size_t)sb * Ec;      // key tile 0
    const __bf16* kb1 = kb0 + (size_t)16 * Ec;           // key tile 1
    bk[0] = join16(*(const v8bf*)(kb0 + 0),  *(const v8bf*)(kb0 + 8));
    bk[1] = join16(*(const v8bf*)(kb0 + 32), *(const v8bf*)(kb0 + 40));
    bk[2] = join16(*(const v8bf*)(kb1 + 0),  *(const v8bf*)(kb1 + 8));
    bk[3] = join16(*(const v8bf*)(kb1 + 32), *(const v8bf*)(kb1 + 40));
  };

  // one 32-key step: consumes bkCur (already resident), preloads bkNext
  auto step = [&](int sb, const v16bf bkCur[4], v16bf bkNext[4]) {
    // V frags for THIS step: issued before the QK WMMAs; the softmax phase
    // hides their latency before the PV WMMAs consume them.
    const __bf16* vb = VT + vRow + sb;
    v16bf bv[4];
#pragma unroll
    for (int dt = 0; dt < 4; ++dt) {
      const __bf16* vp = vb + (size_t)dt * 16 * Sc;
      bv[dt] = join16(*(const v8bf*)vp, *(const v8bf*)(vp + 8));
    }
    // K frags for the NEXT step: a full iteration of latency cover.
    if (sb + 32 < Sc) loadK(sb + 32, bkNext);
    if (sb + 64 < Sc) {
      __builtin_prefetch(K + kRow + (size_t)(sb + 64) * Ec, 0, 1);
      __builtin_prefetch(vb + 64, 0, 1);
    }

    // scores: two 16x16 tiles covering keys [sb, sb+16) and [sb+16, sb+32)
    v8f c0 = {}, c1 = {};
    c0 = wmma_bf16(aq[0], bkCur[0], c0);
    c0 = wmma_bf16(aq[1], bkCur[1], c0);
    c1 = wmma_bf16(aq[0], bkCur[2], c1);
    c1 = wmma_bf16(aq[1], bkCur[3], c1);

    // online softmax over the 32 new keys (row = laneHi*8 + r, col = laneLo)
#pragma unroll
    for (int r = 0; r < 8; ++r) {
      const float mb   = rowmax16(fmaxf(c0[r], c1[r]));  // DPP, no LDS
      const float mn   = fmaxf(mrow[r], mb);
      const float corr = __expf(mrow[r] - mn);
      mrow[r] = mn;
      const float p0 = __expf(c0[r] - mn);
      const float p1 = __expf(c1[r] - mn);
      lrow[r] *= corr;
#pragma unroll
      for (int dt = 0; dt < 4; ++dt) accd[dt][r] *= corr;
      Pbuf[wave][laneHi * 8 + r][laneLo]      = (__bf16)p0;
      Pbuf[wave][laneHi * 8 + r][laneLo + 16] = (__bf16)p1;
    }
    // wave-private LDS: same-wave DS ops are in-order; wait for stores
    asm volatile("s_wait_dscnt 0" ::: "memory");

    // reload P in A-frag layout (lane = M = laneLo, K split per laneHi)
    const __bf16* pb = &Pbuf[wave][laneLo][laneHi * 16];
    v16bf ap = join16(*(const v8bf*)pb, *(const v8bf*)(pb + 8));

    // row sums via matrix op: psum = P @ ones (replicated across cols,
    // lands in the same VGPR-row layout as lrow/accd)
    v8f zero = {};
    v8f psum = wmma_bf16(ap, ones, zero);
#pragma unroll
    for (int r = 0; r < 8; ++r) lrow[r] += psum[r];

    // ctx += P @ V over 4 hd tiles
#pragma unroll
    for (int dt = 0; dt < 4; ++dt) accd[dt] = wmma_bf16(ap, bv[dt], accd[dt]);
  };

  // ping-pong double buffer, 2x unrolled so the swap costs no moves
  v16bf bkA[4], bkB[4];
  loadK(0, bkA);
  for (int sb = 0; sb < Sc; sb += 64) {
    step(sb,      bkA, bkB);
    step(sb + 32, bkB, bkA);
  }

  // normalize and store ctx (bf16, row-major [M, E])
  const size_t oBase =
      ((size_t)b * Tc + qb * 16 + laneHi * 8) * Ec + h * HDc + laneLo;
#pragma unroll
  for (int r = 0; r < 8; ++r) {
    const float inv = 1.0f / lrow[r];
#pragma unroll
    for (int dt = 0; dt < 4; ++dt)
      O[oBase + (size_t)r * Ec + dt * 16] = (__bf16)(accd[dt][r] * inv);
  }
}

}  // namespace

extern "C" void kernel_launch(void* const* d_in, const int* in_sizes, int n_in,
                              void* d_out, int out_size, void* d_ws, size_t ws_size,
                              hipStream_t stream) {
  (void)in_sizes; (void)n_in; (void)out_size; (void)ws_size;
  const float* q  = (const float*)d_in[0];
  const float* k  = (const float*)d_in[1];
  const float* v  = (const float*)d_in[2];
  const float* Wq = (const float*)d_in[3];
  const float* bq = (const float*)d_in[4];
  const float* Wk = (const float*)d_in[5];
  const float* bk = (const float*)d_in[6];
  const float* Wv = (const float*)d_in[7];
  const float* bv = (const float*)d_in[8];
  const float* Wo = (const float*)d_in[9];
  const float* bo = (const float*)d_in[10];

  __bf16* Qp  = (__bf16*)d_ws;                  // [M, E]  bf16 (pre-scaled)
  __bf16* Kp  = Qp  + (size_t)Mc * Ec;          // [M, E]  bf16
  __bf16* VpT = Kp  + (size_t)Mc * Ec;          // [B, H, hd, T] bf16
  __bf16* Ctx = VpT + (size_t)Mc * Ec;          // [M, E]  bf16

  const int tiles = (Mc / 16) * (Ec / 16);      // 32768 tiles, 8 waves/block
  const dim3 pgrid(tiles / 8), pblk(256);
  const float qscale = 0.125f;                  // 1/sqrt(hd), folded into Q proj

  proj_kernel<float, 0><<<pgrid, pblk, 0, stream>>>(q, Wq, bq, Qp, qscale);
  proj_kernel<float, 0><<<pgrid, pblk, 0, stream>>>(k, Wk, bk, Kp, 1.0f);
  proj_kernel<float, 1><<<pgrid, pblk, 0, stream>>>(v, Wv, bv, VpT, 1.0f);

  attn_kernel<<<dim3((Bc * Hc * (Tc / 16)) / 4), dim3(128), 0, stream>>>(
      Qp, Kp, VpT, Ctx);

  proj_kernel<__bf16, 2><<<pgrid, pblk, 0, stream>>>(Ctx, Wo, bo, d_out, 1.0f);
}
